// SparseBlock_24232205484455
// MI455X (gfx1250) — matrix operand order repaired and checked by reading, not compiled, not run on hardware
//
#include <hip/hip_runtime.h>

// ---------------------------------------------------------------------------
// Problem constants: N=2, Cin=64, Cout=256, H=W=512, 32x32 blocks, 256 blocks.
// ---------------------------------------------------------------------------
#define HW     (512 * 512)
#define IMW    512
#define CIN    64
#define COUT   256
#define NBLK   256
#define EPSF   1e-5f
#define HP     34            // padded block edge (32 + 1 halo each side)
#define HPPX   (HP * HP)     // 1156 padded pixels per block

// ws layout (bf16 element offsets)
#define H_ELEMS   ((long)NBLK * HPPX * COUT)        // 75,759,616
#define W2P_OFF   (H_ELEMS)                         // COUT*9*COUT = 589,824
#define W1P_OFF   (W2P_OFF + (long)COUT * 9 * COUT) // COUT*CIN = 16,384
#define XP_OFF    (W1P_OFF + (long)COUT * CIN)      // NBLK*1024*CIN

// conv2 LDS slab: 6 padded rows x 34 cols x 256 ci bf16 = 104,448 B = 13,056 x 8B
#define SLAB_QW   13056

typedef __attribute__((ext_vector_type(2)))  float    v2f;
typedef __attribute__((ext_vector_type(8)))  float    v8f;
typedef __attribute__((ext_vector_type(8)))  __bf16   v8bf;
typedef __attribute__((ext_vector_type(16))) __bf16   v16bf;
typedef __attribute__((ext_vector_type(4)))  unsigned v4u;
typedef __attribute__((ext_vector_type(8)))  int      v8i;
typedef __attribute__((ext_vector_type(4)))  int      v4i;

static __device__ __forceinline__ v16bf cat16(v8bf lo, v8bf hi) {
  return __builtin_shufflevector(lo, hi, 0, 1, 2, 3, 4, 5, 6, 7,
                                 8, 9, 10, 11, 12, 13, 14, 15);
}

// ===========================================================================
// Prep 1: weights -> bf16, operand-major layouts.
// ===========================================================================
__global__ __launch_bounds__(256) void pack_weights_kernel(
    const float* __restrict__ w1, const float* __restrict__ w2,
    __bf16* __restrict__ w1p, __bf16* __restrict__ w2p) {
  const int t0 = blockIdx.x * 256 + threadIdx.x;
  if (t0 < COUT * CIN) w1p[t0] = (__bf16)w1[t0];
  for (int i = t0; i < COUT * 9 * COUT; i += gridDim.x * 256) {
    const int co  = i / (9 * COUT);
    const int rem = i % (9 * COUT);
    const int tap = rem / COUT;
    const int ci  = rem % COUT;
    w2p[i] = (__bf16)w2[(co * COUT + ci) * 9 + tap];
  }
}

// ===========================================================================
// Prep 2: zero the padded h buffer (borders must read as 0 for SAME conv).
// ===========================================================================
__global__ __launch_bounds__(256) void zero_h_kernel(uint4* __restrict__ hz) {
  const long total = H_ELEMS * 2 / 16;
  for (long i = blockIdx.x * 256L + threadIdx.x; i < total;
       i += (long)gridDim.x * 256)
    hz[i] = make_uint4(0u, 0u, 0u, 0u);
}

// ===========================================================================
// Prep 3: gather x blocks -> bf16 xp[blk][px][ci] (channel-contiguous).
// ===========================================================================
__global__ __launch_bounds__(256) void xpack_kernel(
    const float* __restrict__ x, const int* __restrict__ idx,
    __bf16* __restrict__ xp) {
  const int blk = blockIdx.x;
  const int n  = idx[blk * 3 + 0];
  const int gh = idx[blk * 3 + 1];
  const int gw = idx[blk * 3 + 2];
  for (int p = threadIdx.x; p < 1024; p += 256) {
    const int r = p >> 5, c = p & 31;
    const float* xs = x + n * CIN * HW + (gh * 32 + r) * IMW + gw * 32 + c;
    unsigned w[CIN / 2];
    #pragma unroll
    for (int ci = 0; ci < CIN; ci += 2) {
      const unsigned short a =
          __builtin_bit_cast(unsigned short, (__bf16)xs[ci * HW]);
      const unsigned short b =
          __builtin_bit_cast(unsigned short, (__bf16)xs[(ci + 1) * HW]);
      w[ci >> 1] = (unsigned)a | ((unsigned)b << 16);
    }
    uint4* dst = (uint4*)(xp + (((blk << 10) + p) << 6));
    #pragma unroll
    for (int i = 0; i < 8; ++i)
      dst[i] = make_uint4(w[4 * i], w[4 * i + 1], w[4 * i + 2], w[4 * i + 3]);
  }
}

// ===========================================================================
// Kernel A: dense ybase = wd(1x1) * x + bd  (fp32 WMMA 16x16x4).
// Bandwidth-bound (1 GB y stream) -> non-temporal stores keep L2 for x/h.
// ===========================================================================
__global__ __launch_bounds__(256) void dense1x1_kernel(
    const float* __restrict__ x, const float* __restrict__ wd,
    const float* __restrict__ bd, float* __restrict__ y) {
  const int lane = threadIdx.x & 31;
  const int wave = threadIdx.x >> 5;
  const int lrow = lane & 15;
  const int lhi  = lane >> 4;

  const int cb   = blockIdx.y * 16;
  const int fpix = (blockIdx.x * 8 + wave) * 64;
  const int n    = fpix / HW;
  const int p0   = fpix % HW;

  const float* xn = x + n * CIN * HW + p0;
  float*       yn = y + n * COUT * HW + p0;

  v8f acc[4] = {};
  #pragma unroll
  for (int k = 0; k < CIN; k += 4) {
    v2f a;
    a.x = wd[(cb + lrow) * CIN + (k + 0 + 2 * lhi)];
    a.y = wd[(cb + lrow) * CIN + (k + 1 + 2 * lhi)];
    #pragma unroll
    for (int t = 0; t < 4; ++t) {
      const float* xpx = xn + (k + 2 * lhi) * HW + t * 16 + lrow;
      v2f b;
      b.x = xpx[0];
      b.y = xpx[HW];
      acc[t] = __builtin_amdgcn_wmma_f32_16x16x4_f32(
          false, a, false, b, (short)0, acc[t], false, false);
    }
  }
  #pragma unroll
  for (int t = 0; t < 4; ++t) {
    #pragma unroll
    for (int r = 0; r < 8; ++r) {
      const int co = cb + r + 8 * lhi;
      __builtin_nontemporal_store(acc[t][r] + bd[co],
                                  &yn[co * HW + t * 16 + lrow]);
    }
  }
}

// ===========================================================================
// Kernel B: conv1(1x1, 64->256) + BN + ReLU from packed xp, write bf16 h
// into the PADDED buffer interior.  Epilogue packs bf16 pairs -> b32 stores.
// ===========================================================================
__global__ __launch_bounds__(256) void conv1_bn_relu_kernel(
    const __bf16* __restrict__ xp, const __bf16* __restrict__ w1p,
    const float* __restrict__ b1, const float* __restrict__ g1,
    const float* __restrict__ be1, const float* __restrict__ m1,
    const float* __restrict__ v1, __bf16* __restrict__ hbuf) {
  const int lane = threadIdx.x & 31;
  const int wave = threadIdx.x >> 5;
  const int lrow = lane & 15;
  const int lhi  = lane >> 4;

  const int blk   = blockIdx.z;
  const int cb    = blockIdx.y * 16;
  const int pbase = (blockIdx.x * 8 + wave) * 64;

  v8f acc[4] = {};
  #pragma unroll
  for (int k0 = 0; k0 < CIN; k0 += 32) {
    const __bf16* wa = w1p + (cb + lrow) * CIN + k0 + 8 * lhi;
    const v16bf A = cat16(*(const v8bf*)wa, *(const v8bf*)(wa + 16));
    #pragma unroll
    for (int t = 0; t < 4; ++t) {
      const int p = pbase + t * 16 + lrow;
      const __bf16* xb = xp + (((blk << 10) + p) << 6) + k0 + 8 * lhi;
      const v16bf B = cat16(*(const v8bf*)xb, *(const v8bf*)(xb + 16));
      acc[t] = __builtin_amdgcn_wmma_f32_16x16x32_bf16(
          false, A, false, B, (short)0, acc[t], false, false);
    }
  }

  // per-co BN affine (8 co values owned by this lane)
  float s[8], sh[8];
  #pragma unroll
  for (int rr = 0; rr < 8; ++rr) {
    const int co = cb + rr + 8 * lhi;
    s[rr]  = g1[co] / sqrtf(v1[co] + EPSF);
    sh[rr] = be1[co] - m1[co] * s[rr] + b1[co] * s[rr];
  }
  #pragma unroll
  for (int t = 0; t < 4; ++t) {
    const int p = pbase + t * 16 + lrow;
    const int r = p >> 5, c = p & 31;
    __bf16* dst =
        hbuf + ((blk * HPPX + (r + 1) * HP + (c + 1)) << 8) + cb + 8 * lhi;
    #pragma unroll
    for (int rp = 0; rp < 4; ++rp) {
      float v0 = acc[t][2 * rp + 0] * s[2 * rp + 0] + sh[2 * rp + 0];
      float v1f = acc[t][2 * rp + 1] * s[2 * rp + 1] + sh[2 * rp + 1];
      v0  = v0  > 0.f ? v0  : 0.f;
      v1f = v1f > 0.f ? v1f : 0.f;
      const unsigned pk =
          (unsigned)__builtin_bit_cast(unsigned short, (__bf16)v0) |
          ((unsigned)__builtin_bit_cast(unsigned short, (__bf16)v1f) << 16);
      ((unsigned*)dst)[rp] = pk;
    }
  }
}

// ===========================================================================
// Kernel C (hot: 309 GFLOP): conv2 3x3 SAME + BN + ReLU + scatter into y.
// WG macro-tile: 128 cout x 128 px.  The 104 KB h slab is contiguous in the
// padded layout -> staged via ONE Tensor-Data-Mover DMA (tensor_load_to_lds,
// TENSORcnt) issued by wave 0; fallback is a vector copy.  B operands are
// ds_load_b128 pairs; each B feeds 2 WMMAs.
// ===========================================================================
__global__ __launch_bounds__(256) void conv2_bn_relu_scatter_kernel(
    const __bf16* __restrict__ hbuf, const __bf16* __restrict__ w2p,
    const int* __restrict__ idx, const float* __restrict__ b2,
    const float* __restrict__ g2, const float* __restrict__ be2,
    const float* __restrict__ m2, const float* __restrict__ v2,
    float* __restrict__ y) {
  __shared__ __bf16 hs[6 * HP * COUT];   // 104,448 bytes

  const int lane = threadIdx.x & 31;
  const int wave = threadIdx.x >> 5;
  const int lrow = lane & 15;
  const int lhi  = lane >> 4;
  const int wco  = wave & 3;             // 4 cout groups of 32
  const int wpx  = wave >> 2;            // 2 px groups of 64

  const int blk   = blockIdx.z;
  const int cbw   = blockIdx.y * 128 + wco * 32;       // wave cout base
  const int rbase = blockIdx.x * 4;                    // 4 rows per WG
  const int pbase = rbase * 32 + wpx * 64;             // wave px base

  const int n  = idx[blk * 3 + 0];
  const int gh = idx[blk * 3 + 1];
  const int gw = idx[blk * 3 + 2];

  const __bf16* slab = hbuf + (((long)blk * HPPX + rbase * HP) << 8);

#if defined(__HIP_DEVICE_COMPILE__) && \
    __has_builtin(__builtin_amdgcn_tensor_load_to_lds) && \
    __has_builtin(__builtin_amdgcn_s_wait_tensorcnt)
  // ---- TDM: one DMA for the whole slab (13,056 x 8B, 1-D tile) ----
  if (wave == 0) {
    const unsigned long long ga = (unsigned long long)(size_t)slab;
    const unsigned lds_off = (unsigned)(size_t)(void*)hs;  // LDS aperture low32
    v4u g0;
    g0[0] = 1u;                                    // count=1, user descriptor
    g0[1] = lds_off;                               // lds_addr
    g0[2] = (unsigned)(ga & 0xFFFFFFFFu);          // global_addr[31:0]
    g0[3] = (unsigned)((ga >> 32) & 0x01FFFFFFu)   // global_addr[56:32]
            | (2u << 30);                          // type = 2 ("image")
    v8i g1;
    g1[0] = 0x00030000;                // data_size=3 (8B), no multicast/pad
    g1[1] = (int)((unsigned)SLAB_QW << 16);        // tensor_dim0[15:0]
    g1[2] = (int)(1u << 16);                       // tensor_dim1 = 1
    g1[3] = (int)((unsigned)SLAB_QW << 16);        // tile_dim0
    g1[4] = 1;                                     // tile_dim1 = 1
    g1[5] = SLAB_QW;                               // tensor_dim0_stride
    g1[6] = 0;
    g1[7] = 0;
    v4i g2 = {0, 0, 0, 0}, g3 = {0, 0, 0, 0};      // <=2D: groups 2/3 unused
    v8i g4 = {0, 0, 0, 0, 0, 0, 0, 0};             // trailing group (unused)
    __builtin_amdgcn_tensor_load_to_lds(g0, g1, g2, g3, g4, 0);
    __builtin_amdgcn_s_wait_tensorcnt(0);
  }
  __syncthreads();
#else
  // ---- fallback: cooperative vector copy ----
  {
    const uint4* src = (const uint4*)slab;
    uint4* dst = (uint4*)hs;
    #pragma unroll 2
    for (int i = threadIdx.x; i < 6 * HP * COUT / 8; i += 256) dst[i] = src[i];
  }
  __syncthreads();
#endif

  v8f acc[2][4] = {};

  const __bf16* w0 = w2p + (cbw + lrow) * 9 * COUT + 8 * lhi;  // co tile 0
  const __bf16* w1r = w0 + 16 * 9 * COUT;                      // co tile 1

  for (int dr = -1; dr <= 1; ++dr) {
    for (int dc = -1; dc <= 1; ++dc) {
      const int tap = (dr + 1) * 3 + (dc + 1);
      int soff[4];
      #pragma unroll
      for (int t = 0; t < 4; ++t) {
        const int p  = pbase + t * 16;
        const int lr = (p >> 5) - rbase + dr + 1;      // slab row 0..5
        soff[t] = ((lr * HP + (p & 31) + lrow + dc + 1) << 8) + 8 * lhi;
      }
      #pragma unroll
      for (int k0 = 0; k0 < COUT; k0 += 32) {
        const __bf16* wa0 = w0 + tap * COUT + k0;
        const __bf16* wa1 = w1r + tap * COUT + k0;
        const v16bf A0 = cat16(*(const v8bf*)wa0, *(const v8bf*)(wa0 + 16));
        const v16bf A1 = cat16(*(const v8bf*)wa1, *(const v8bf*)(wa1 + 16));
        #pragma unroll
        for (int t = 0; t < 4; ++t) {
          const __bf16* hp = hs + soff[t] + k0;
          const v16bf B = cat16(*(const v8bf*)hp, *(const v8bf*)(hp + 16));
          acc[0][t] = __builtin_amdgcn_wmma_f32_16x16x32_bf16(
              false, A0, false, B, (short)0, acc[0][t], false, false);
          acc[1][t] = __builtin_amdgcn_wmma_f32_16x16x32_bf16(
              false, A1, false, B, (short)0, acc[1][t], false, false);
        }
      }
    }
  }

  #pragma unroll
  for (int j = 0; j < 2; ++j) {
    #pragma unroll
    for (int rr = 0; rr < 8; ++rr) {
      const int co = cbw + 16 * j + rr + 8 * lhi;
      const float s  = g2[co] / sqrtf(v2[co] + EPSF);
      const float sh = be2[co] - m2[co] * s + b2[co] * s;
      #pragma unroll
      for (int t = 0; t < 4; ++t) {
        const int p = pbase + t * 16 + lrow;
        const int r = p >> 5, c = p & 31;
        float val = acc[j][t][rr] * s + sh;
        val = val > 0.f ? val : 0.f;
        __builtin_nontemporal_store(
            val, &y[(n * COUT + co) * HW + (gh * 32 + r) * IMW + gw * 32 + c]);
      }
    }
  }
}

// ===========================================================================
// Launch.
// ===========================================================================
extern "C" void kernel_launch(void* const* d_in, const int* in_sizes, int n_in,
                              void* d_out, int out_size, void* d_ws, size_t ws_size,
                              hipStream_t stream) {
  (void)in_sizes; (void)n_in; (void)out_size; (void)ws_size;
  const float* x       = (const float*)d_in[0];
  const int*   indices = (const int*)d_in[2];
  const float* w1 = (const float*)d_in[3];
  const float* b1 = (const float*)d_in[4];
  const float* g1 = (const float*)d_in[5];
  const float* be1 = (const float*)d_in[6];
  const float* m1 = (const float*)d_in[7];
  const float* v1 = (const float*)d_in[8];
  const float* w2 = (const float*)d_in[9];
  const float* b2 = (const float*)d_in[10];
  const float* g2 = (const float*)d_in[11];
  const float* be2 = (const float*)d_in[12];
  const float* m2 = (const float*)d_in[13];
  const float* v2 = (const float*)d_in[14];
  const float* wd = (const float*)d_in[21];
  const float* bd = (const float*)d_in[22];

  float*  y    = (float*)d_out;
  __bf16* ws   = (__bf16*)d_ws;
  __bf16* hbuf = ws;
  __bf16* w2p  = ws + W2P_OFF;
  __bf16* w1p  = ws + W1P_OFF;
  __bf16* xp   = ws + XP_OFF;

  pack_weights_kernel<<<dim3(2304), dim3(256), 0, stream>>>(w1, w2, w1p, w2p);
  zero_h_kernel<<<dim3(4624), dim3(256), 0, stream>>>((uint4*)hbuf);
  xpack_kernel<<<dim3(NBLK), dim3(256), 0, stream>>>(x, indices, xp);

  // dense base must land in y before the sparse scatter overwrites blocks
  dense1x1_kernel<<<dim3(1024, 16, 1), dim3(256), 0, stream>>>(x, wd, bd, y);

  conv1_bn_relu_kernel<<<dim3(2, 16, NBLK), dim3(256), 0, stream>>>(
      xp, w1p, b1, g1, be1, m1, v1, hbuf);

  // 8 row-groups x 2 cout-groups x 256 blocks; 104 KB LDS per WG.
  conv2_bn_relu_scatter_kernel<<<dim3(8, 2, NBLK), dim3(256), 0, stream>>>(
      hbuf, w2p, indices, b2, g2, be2, m2, v2, y);
}
